// MultiHeadSelfAttention_53034256171582
// MI455X (gfx1250) — matrix operand order repaired and checked by reading, not compiled, not run on hardware
//
#include <hip/hip_runtime.h>

// MI455X (gfx1250) multi-head attention forward, f16 WMMA compute / f32 accumulate.
// Pipeline: convert(fp32->f16, deinterleave qkv weights) -> qkv GEMM -> flash attention -> proj GEMM.
// Data movement uses GLOBAL_LOAD_ASYNC_TO_LDS_B128 (ASYNCcnt) with 2-stage LDS double buffering.

typedef _Float16 v16h __attribute__((ext_vector_type(16)));
typedef _Float16 v8h  __attribute__((ext_vector_type(8)));
typedef float    v8f  __attribute__((ext_vector_type(8)));
typedef int      v4i  __attribute__((ext_vector_type(4)));

namespace {
constexpr int  Bc   = 2;
constexpr int  Nc   = 2048;
constexpr int  Ec   = 1024;
constexpr int  Hc   = 16;
constexpr int  Dc   = 64;
constexpr int  Mtot = Bc * Nc;            // 4096 tokens
constexpr size_t SZ_XH = (size_t)Mtot * Ec;      // 4M halfs
constexpr size_t SZ_W  = (size_t)Ec * Ec;        // 1M halfs
constexpr size_t SZ_T  = (size_t)Bc * Hc * Nc * Dc; // 4M halfs (Q/K/Vt each)
// workspace byte offsets (total 48 MiB)
constexpr size_t OFF_XH = 0;
constexpr size_t OFF_WQ = OFF_XH + SZ_XH * 2;
constexpr size_t OFF_WK = OFF_WQ + SZ_W * 2;
constexpr size_t OFF_WV = OFF_WK + SZ_W * 2;
constexpr size_t OFF_WP = OFF_WV + SZ_W * 2;
constexpr size_t OFF_Q  = OFF_WP + SZ_W * 2;
constexpr size_t OFF_K  = OFF_Q  + SZ_T * 2;
constexpr size_t OFF_VT = OFF_K  + SZ_T * 2;
constexpr size_t OFF_O  = OFF_VT + SZ_T * 2;
}

// ---- async global->LDS copy (16B per lane), with sync fallback ----
__device__ __forceinline__ void async_copy16(const _Float16* g, _Float16* l) {
#if defined(__HIP_DEVICE_COMPILE__) && \
    __has_builtin(__builtin_amdgcn_global_load_async_to_lds_b128)
  typedef __attribute__((address_space(1))) v4i* gv4i_p;
  typedef __attribute__((address_space(3))) v4i* lv4i_p;
  __builtin_amdgcn_global_load_async_to_lds_b128(
      (gv4i_p)(uintptr_t)g, (lv4i_p)(uint32_t)(uintptr_t)l, 0, 0);
#else
  *(v8h*)l = *(const v8h*)g;
#endif
}

__device__ __forceinline__ void wait_async_and_barrier() {
#if defined(__HIP_DEVICE_COMPILE__) && \
    __has_builtin(__builtin_amdgcn_s_wait_asynccnt)
  __builtin_amdgcn_s_wait_asynccnt(0);
#elif defined(__HIP_DEVICE_COMPILE__)
  asm volatile("s_wait_asynccnt 0" ::: "memory");
#endif
  __syncthreads();
}

// ---- WMMA fragment helpers (layouts per CDNA5 ISA 7.12.2, wave32) ----

// A-matrix 16x32 f16: lane r=lane&15 is row M; hi=lane>>4 selects K subsets
// elements 0..7 -> K = kbase + hi*8 + j ; elements 8..15 -> K = kbase + 16 + hi*8 + j
__device__ __forceinline__ v16h load_fragA(const _Float16* tile, int ld, int kbase) {
  const int lane = threadIdx.x & 31;
  const int r = lane & 15, hi = lane >> 4;
  const _Float16* p = tile + (size_t)r * ld + kbase + hi * 8;
  union { v16h v; v8h h[2]; } u;
  u.h[0] = *(const v8h*)(p);
  u.h[1] = *(const v8h*)(p + 16);
  return u.v;
}

// B-matrix 32x16 f16: lane n=lane&15 is column N; elements 0..7 -> K = kbase + hi*8 + t ; elements 8..15 -> K = kbase + 16 + hi*8 + (t-8)
__device__ __forceinline__ v16h load_fragB(const _Float16* tile, int ld, int kbase) {
  const int lane = threadIdx.x & 31;
  const int n = lane & 15, hi = lane >> 4;
  const _Float16* p = tile + (size_t)n * ld + kbase + hi * 8;
  union { v16h v; v8h h[2]; } u;
  u.h[0] = *(const v8h*)(p);
  u.h[1] = *(const v8h*)(p + 16);
  return u.v;
}

__device__ __forceinline__ v8f wmma_f16(v16h a, v16h b, v8f c) {
  return __builtin_amdgcn_wmma_f32_16x16x32_f16(false, a, false, b, (short)0, c,
                                                false, false);
}

// ---- Kernel 1: fp32 -> f16 conversion + QKV weight de-interleave ----
// w_qkv row for output feature n (n = h*64+d) of component c is 3*n + c.
__global__ __launch_bounds__(256) void convert_kernel(
    const float* __restrict__ x, const float* __restrict__ wqkv,
    const float* __restrict__ wproj,
    _Float16* __restrict__ xh, _Float16* __restrict__ wq,
    _Float16* __restrict__ wk, _Float16* __restrict__ wv,
    _Float16* __restrict__ wp) {
  size_t i = (size_t)blockIdx.x * 256 + threadIdx.x;
  const size_t M4 = SZ_XH;
  if (i < M4) {
    xh[i] = (_Float16)x[i];
    return;
  }
  size_t r = i - M4;
  if (r < SZ_W) {               // Wq
    size_t row = r >> 10, k = r & 1023;
    wq[r] = (_Float16)wqkv[(3 * row + 0) * Ec + k];
  } else if (r < 2 * SZ_W) {    // Wk
    r -= SZ_W;
    size_t row = r >> 10, k = r & 1023;
    wk[r] = (_Float16)wqkv[(3 * row + 1) * Ec + k];
  } else if (r < 3 * SZ_W) {    // Wv
    r -= 2 * SZ_W;
    size_t row = r >> 10, k = r & 1023;
    wv[r] = (_Float16)wqkv[(3 * row + 2) * Ec + k];
  } else if (r < 4 * SZ_W) {    // Wproj
    r -= 3 * SZ_W;
    wp[r] = (_Float16)wproj[r];
  }
}

// ---- Kernel 2: QKV GEMM  C[4096 x 1024] = Xh @ Wc^T + b, scatter to Q/K/Vt ----
// 128x128 block tile, 8 waves as 4(M) x 2(N), 32x64 wave tile, 2-stage async pipeline.
// blockIdx.z = component (0=Q scaled by 1/sqrt(D), 1=K, 2=V stored transposed)
__global__ __launch_bounds__(256) void qkv_gemm_kernel(
    const _Float16* __restrict__ xh, const _Float16* __restrict__ wq,
    const _Float16* __restrict__ wk, const _Float16* __restrict__ wv,
    const float* __restrict__ bqkv, _Float16* __restrict__ Qh,
    _Float16* __restrict__ Kh, _Float16* __restrict__ Vt) {
  __shared__ __align__(16) _Float16 As[2][128 * 32];
  __shared__ __align__(16) _Float16 Bs[2][128 * 32];
  const int c = blockIdx.z;
  const _Float16* w = (c == 0) ? wq : ((c == 1) ? wk : wv);
  const int m0 = blockIdx.x * 128, n0 = blockIdx.y * 128;
  const int tid = threadIdx.x, wave = tid >> 5, lane = tid & 31, hi = lane >> 4;
  const int wm = (wave >> 1) * 32, wn = (wave & 1) * 64;

  auto stage = [&](int s) {
    const int buf = s & 1;
    const int kt = s * 32;
#pragma unroll
    for (int it = 0; it < 2; ++it) {  // A: 128x32 halfs = 512 x 16B
      int v = tid + it * 256;
      int row = v >> 2, col8 = (v & 3) * 8;
      async_copy16(&xh[(size_t)(m0 + row) * Ec + kt + col8],
                   &As[buf][row * 32 + col8]);
    }
#pragma unroll
    for (int it = 0; it < 2; ++it) {  // B: 128x32 halfs
      int v = tid + it * 256;
      int row = v >> 2, col8 = (v & 3) * 8;
      async_copy16(&w[(size_t)(n0 + row) * Ec + kt + col8],
                   &Bs[buf][row * 32 + col8]);
    }
  };

  v8f acc[2][4] = {};
  constexpr int NS = Ec / 32;
  stage(0);
  for (int s = 0; s < NS; ++s) {
    wait_async_and_barrier();          // stage s resident, all waves past compute s-1
    if (s + 1 < NS) stage(s + 1);      // overlap next copy with compute
    const int buf = s & 1;
    v16h a0 = load_fragA(&As[buf][(wm + 0) * 32], 32, 0);
    v16h a1 = load_fragA(&As[buf][(wm + 16) * 32], 32, 0);
    v16h b[4];
#pragma unroll
    for (int j = 0; j < 4; ++j)
      b[j] = load_fragB(&Bs[buf][(wn + 16 * j) * 32], 32, 0);
#pragma unroll
    for (int j = 0; j < 4; ++j) {
      acc[0][j] = wmma_f16(a0, b[j], acc[0][j]);
      acc[1][j] = wmma_f16(a1, b[j], acc[1][j]);
    }
  }

#pragma unroll
  for (int i = 0; i < 2; ++i)
#pragma unroll
    for (int j = 0; j < 4; ++j) {
      const int col = n0 + wn + 16 * j + (lane & 15);
      const float bias = bqkv[3 * col + c];
      const int h = col >> 6, d = col & 63;
#pragma unroll
      for (int e = 0; e < 8; ++e) {
        const int row = m0 + wm + 16 * i + e + 8 * hi;
        const int bb = row >> 11;
        const int tok = row & (Nc - 1);
        float val = acc[i][j][e] + bias;
        const size_t bh = (size_t)bb * Hc + h;
        if (c == 0)
          Qh[(bh * Nc + tok) * Dc + d] = (_Float16)(val * 0.125f);
        else if (c == 1)
          Kh[(bh * Nc + tok) * Dc + d] = (_Float16)val;
        else
          Vt[(bh * Dc + d) * Nc + tok] = (_Float16)val;
      }
    }
}

// ---- Kernel 3: flash attention. One workgroup = 128 Q rows of one (b,h). ----
__global__ __launch_bounds__(256) void attn_kernel(
    const _Float16* __restrict__ Qh, const _Float16* __restrict__ Kh,
    const _Float16* __restrict__ Vt, _Float16* __restrict__ Oh) {
  __shared__ __align__(16) _Float16 Ks[2][64 * 64];    // K block  [kr][d]
  __shared__ __align__(16) _Float16 Vs[2][64 * 64];    // Vt block [d][kr]
  __shared__ __align__(16) _Float16 Ps[8 * 16 * 64];   // per-wave P

  const int bh = blockIdx.x >> 4;   // b*H + h
  const int qblk = blockIdx.x & 15;
  const int b = bh >> 4, h = bh & 15;
  const _Float16* Qbase = Qh + (size_t)bh * Nc * Dc;
  const _Float16* Kbase = Kh + (size_t)bh * Nc * Dc;
  const _Float16* Vbase = Vt + (size_t)bh * Dc * Nc;

  const int tid = threadIdx.x, wave = tid >> 5, lane = tid & 31, hi = lane >> 4;
  const int q0 = qblk * 128 + wave * 16;  // this wave's 16 Q rows

  v16h qf[2];
  qf[0] = load_fragA(Qbase + (size_t)q0 * Dc, Dc, 0);
  qf[1] = load_fragA(Qbase + (size_t)q0 * Dc, Dc, 32);

  auto stageKV = [&](int jb) {
    const int buf = jb & 1;
    const int j0 = jb * 64;
#pragma unroll
    for (int it = 0; it < 2; ++it) {  // 64x64 halfs each = 512 x 16B
      int v = tid + it * 256;
      int row = v >> 3, col8 = (v & 7) * 8;
      async_copy16(&Kbase[(size_t)(j0 + row) * Dc + col8],
                   &Ks[buf][row * 64 + col8]);
      async_copy16(&Vbase[(size_t)row * Nc + j0 + col8],
                   &Vs[buf][row * 64 + col8]);
    }
  };

  v8f Oacc[4] = {};
  float mrow[8], lrow[8];
#pragma unroll
  for (int e = 0; e < 8; ++e) { mrow[e] = -1e30f; lrow[e] = 0.0f; }
  _Float16* Pw = Ps + wave * 16 * 64;

  constexpr int NB = Nc / 64;  // 32 K/V blocks
  stageKV(0);
  for (int jb = 0; jb < NB; ++jb) {
    wait_async_and_barrier();
    if (jb + 1 < NB) stageKV(jb + 1);
    const int buf = jb & 1;

    // S = Q @ K^T  (16 x 64), 4 frags x 2 K-steps
    v8f s[4];
#pragma unroll
    for (int jn = 0; jn < 4; ++jn) {
      s[jn] = (v8f){};
#pragma unroll
      for (int ks = 0; ks < 2; ++ks)
        s[jn] = wmma_f16(qf[ks],
                         load_fragB(&Ks[buf][(jn * 16) * 64], 64, ks * 32),
                         s[jn]);
    }

    // online softmax: row lives across 16 lanes of one half-wave
    float alpha[8];
#pragma unroll
    for (int e = 0; e < 8; ++e) {
      float rmax = s[0][e];
#pragma unroll
      for (int jn = 1; jn < 4; ++jn) rmax = fmaxf(rmax, s[jn][e]);
      rmax = fmaxf(rmax, __shfl_xor(rmax, 1, 32));
      rmax = fmaxf(rmax, __shfl_xor(rmax, 2, 32));
      rmax = fmaxf(rmax, __shfl_xor(rmax, 4, 32));
      rmax = fmaxf(rmax, __shfl_xor(rmax, 8, 32));
      float mnew = fmaxf(mrow[e], rmax);
      alpha[e] = __expf(mrow[e] - mnew);
      mrow[e] = mnew;
#pragma unroll
      for (int jn = 0; jn < 4; ++jn) s[jn][e] = __expf(s[jn][e] - mnew);
      float rsum = s[0][e] + s[1][e] + s[2][e] + s[3][e];
      rsum += __shfl_xor(rsum, 1, 32);
      rsum += __shfl_xor(rsum, 2, 32);
      rsum += __shfl_xor(rsum, 4, 32);
      rsum += __shfl_xor(rsum, 8, 32);
      lrow[e] = lrow[e] * alpha[e] + rsum;
#pragma unroll
      for (int jd = 0; jd < 4; ++jd) Oacc[jd][e] *= alpha[e];
    }

    // P (C-layout) -> per-wave LDS -> reload as A fragments
#pragma unroll
    for (int jn = 0; jn < 4; ++jn)
#pragma unroll
      for (int e = 0; e < 8; ++e)
        Pw[(e + 8 * hi) * 64 + jn * 16 + (lane & 15)] = (_Float16)s[jn][e];
    asm volatile("s_wait_dscnt 0" ::: "memory");  // wave-local LDS RAW

    // O += P @ V
#pragma unroll
    for (int ks = 0; ks < 2; ++ks) {
      v16h pa = load_fragA(Pw, 64, ks * 32);
#pragma unroll
      for (int jd = 0; jd < 4; ++jd)
        Oacc[jd] = wmma_f16(
            pa, load_fragB(&Vs[buf][(jd * 16) * 64], 64, ks * 32), Oacc[jd]);
    }
  }

  float invl[8];
#pragma unroll
  for (int e = 0; e < 8; ++e) invl[e] = 1.0f / lrow[e];
#pragma unroll
  for (int jd = 0; jd < 4; ++jd) {
    const int col = h * Dc + jd * 16 + (lane & 15);
#pragma unroll
    for (int e = 0; e < 8; ++e) {
      const int row = q0 + e + 8 * hi;  // token within batch
      Oh[((size_t)b * Nc + row) * Ec + col] = (_Float16)(Oacc[jd][e] * invl[e]);
    }
  }
}

// ---- Kernel 4: output projection  out[4096x1024] = Oh @ Wp^T + b_proj (f32) ----
__global__ __launch_bounds__(256) void proj_gemm_kernel(
    const _Float16* __restrict__ Oh, const _Float16* __restrict__ wp,
    const float* __restrict__ bproj, float* __restrict__ out) {
  __shared__ __align__(16) _Float16 As[2][128 * 32];
  __shared__ __align__(16) _Float16 Bs[2][128 * 32];
  const int m0 = blockIdx.x * 128, n0 = blockIdx.y * 128;
  const int tid = threadIdx.x, wave = tid >> 5, lane = tid & 31, hi = lane >> 4;
  const int wm = (wave >> 1) * 32, wn = (wave & 1) * 64;

  auto stage = [&](int s) {
    const int buf = s & 1;
    const int kt = s * 32;
#pragma unroll
    for (int it = 0; it < 2; ++it) {
      int v = tid + it * 256;
      int row = v >> 2, col8 = (v & 3) * 8;
      async_copy16(&Oh[(size_t)(m0 + row) * Ec + kt + col8],
                   &As[buf][row * 32 + col8]);
    }
#pragma unroll
    for (int it = 0; it < 2; ++it) {
      int v = tid + it * 256;
      int row = v >> 2, col8 = (v & 3) * 8;
      async_copy16(&wp[(size_t)(n0 + row) * Ec + kt + col8],
                   &Bs[buf][row * 32 + col8]);
    }
  };

  v8f acc[2][4] = {};
  constexpr int NS = Ec / 32;
  stage(0);
  for (int s = 0; s < NS; ++s) {
    wait_async_and_barrier();
    if (s + 1 < NS) stage(s + 1);
    const int buf = s & 1;
    v16h a0 = load_fragA(&As[buf][(wm + 0) * 32], 32, 0);
    v16h a1 = load_fragA(&As[buf][(wm + 16) * 32], 32, 0);
    v16h b[4];
#pragma unroll
    for (int j = 0; j < 4; ++j)
      b[j] = load_fragB(&Bs[buf][(wn + 16 * j) * 32], 32, 0);
#pragma unroll
    for (int j = 0; j < 4; ++j) {
      acc[0][j] = wmma_f16(a0, b[j], acc[0][j]);
      acc[1][j] = wmma_f16(a1, b[j], acc[1][j]);
    }
  }

#pragma unroll
  for (int i = 0; i < 2; ++i)
#pragma unroll
    for (int j = 0; j < 4; ++j) {
      const int col = n0 + wn + 16 * j + (lane & 15);
      const float bias = bproj[col];
#pragma unroll
      for (int e = 0; e < 8; ++e) {
        const int row = m0 + wm + 16 * i + e + 8 * hi;
        out[(size_t)row * Ec + col] = acc[i][j][e] + bias;
      }
    }
}

extern "C" void kernel_launch(void* const* d_in, const int* in_sizes, int n_in,
                              void* d_out, int out_size, void* d_ws,
                              size_t ws_size, hipStream_t stream) {
  const float* x     = (const float*)d_in[0];
  const float* wqkv  = (const float*)d_in[1];
  const float* bqkv  = (const float*)d_in[2];
  const float* wproj = (const float*)d_in[3];
  const float* bproj = (const float*)d_in[4];
  float* out = (float*)d_out;

  char* ws = (char*)d_ws;
  _Float16* xh = (_Float16*)(ws + OFF_XH);
  _Float16* wq = (_Float16*)(ws + OFF_WQ);
  _Float16* wk = (_Float16*)(ws + OFF_WK);
  _Float16* wv = (_Float16*)(ws + OFF_WV);
  _Float16* wp = (_Float16*)(ws + OFF_WP);
  _Float16* Qh = (_Float16*)(ws + OFF_Q);
  _Float16* Kh = (_Float16*)(ws + OFF_K);
  _Float16* Vt = (_Float16*)(ws + OFF_VT);
  _Float16* Oh = (_Float16*)(ws + OFF_O);

  // 1) convert: 4M (x) + 3*1M (qkv weights) + 1M (proj weight) = 8M elements
  convert_kernel<<<(8u << 20) / 256, 256, 0, stream>>>(x, wqkv, wproj, xh, wq,
                                                       wk, wv, wp);
  // 2) QKV gemm: M=4096, N=1024 per component, 128x128 tiles
  dim3 g2(Mtot / 128, Ec / 128, 3);
  qkv_gemm_kernel<<<g2, 256, 0, stream>>>(xh, wq, wk, wv, bqkv, Qh, Kh, Vt);
  // 3) attention: (B*H) * (N/128) workgroups
  attn_kernel<<<Bc * Hc * (Nc / 128), 256, 0, stream>>>(Qh, Kh, Vt, Oh);
  // 4) projection
  dim3 g3(Mtot / 128, Ec / 128, 1);
  proj_gemm_kernel<<<g3, 256, 0, stream>>>(Oh, wp, bproj, out);
}
